// MultiHeadSelfAttention_82643760710104
// MI455X (gfx1250) — matrix operand order repaired and checked by reading, not compiled, not run on hardware
//
#include <hip/hip_runtime.h>
#include <hip/hip_bf16.h>

// MI455X / gfx1250, wave32. All matmuls via v_wmma_f32_16x16x32_bf16.
#define B_  2
#define S_  2048
#define D_  1024
#define H_  16
#define HD_ 64
#define M_  (B_ * S_)   // 4096 rows of (b,s)

typedef __bf16 bf16_t;
typedef __attribute__((ext_vector_type(8)))  __bf16 v8bf;
typedef __attribute__((ext_vector_type(16))) __bf16 v16bf;
typedef __attribute__((ext_vector_type(8)))  float  v8f;

static __device__ __forceinline__ bf16_t to_bf16(float f) {
  unsigned u = __builtin_bit_cast(unsigned, f);
  u += 0x7FFFu + ((u >> 16) & 1u);              // round-to-nearest-even
  unsigned short h = (unsigned short)(u >> 16);
  return __builtin_bit_cast(bf16_t, h);
}

static __device__ __forceinline__ unsigned pack2(float lo, float hi) {
  unsigned a = __builtin_bit_cast(unsigned, lo);
  a += 0x7FFFu + ((a >> 16) & 1u);
  unsigned b = __builtin_bit_cast(unsigned, hi);
  b += 0x7FFFu + ((b >> 16) & 1u);
  return (a >> 16) | (b & 0xFFFF0000u);
}

// Build a v16bf operand from two 16-byte aligned chunks of 8 bf16 each.
static __device__ __forceinline__ v16bf ldfrag(const bf16_t* p0, const bf16_t* p1) {
  v8bf lo = *(const v8bf*)p0;
  v8bf hi = *(const v8bf*)p1;
  return __builtin_shufflevector(lo, hi, 0,1,2,3,4,5,6,7,8,9,10,11,12,13,14,15);
}

static __device__ __forceinline__ v8f wmma_bf16(v16bf a, v16bf b, v8f c) {
  // D = A(16x32) * B(32x16) + C(16x16 f32)
  return __builtin_amdgcn_wmma_f32_16x16x32_bf16(false, a, false, b, (short)0, c,
                                                 false, false);
}

// ---------------------------------------------------------------------------
// Kernel 1: convert f32 inputs -> bf16 workspace copies
// ---------------------------------------------------------------------------
__global__ void k_prep(const float* __restrict__ X,  const float* __restrict__ Wq,
                       const float* __restrict__ Wk, const float* __restrict__ Wv,
                       const float* __restrict__ Wo,
                       bf16_t* __restrict__ Xb,  bf16_t* __restrict__ Wqb,
                       bf16_t* __restrict__ Wkb, bf16_t* __restrict__ Wvb,
                       bf16_t* __restrict__ Wob) {
  long i = (long)blockIdx.x * blockDim.x + threadIdx.x;
  const long nx = (long)M_ * D_;
  const long nw = (long)D_ * D_;
  if (i < nx) { Xb[i]  = to_bf16(X[i]);  return; }
  i -= nx;
  if (i < nw) { Wqb[i] = to_bf16(Wq[i]); return; }
  i -= nw;
  if (i < nw) { Wkb[i] = to_bf16(Wk[i]); return; }
  i -= nw;
  if (i < nw) { Wvb[i] = to_bf16(Wv[i]); return; }
  i -= nw;
  if (i < nw) { Wob[i] = to_bf16(Wo[i]); }
}

// ---------------------------------------------------------------------------
// Kernel 2: QKV projection (X @ W^T) + fused RoPE on Q,K.  One wave computes a
// 32(row) x 64(one head) tile for one of {Q,K,V}: 2 A-frags x 4 B-frags ->
// 8 WMMAs per 32-k step (double register-level A/B reuse vs 16-row tiles).
// Q,K stored [bh,S,64] bf16 (Q pre-scaled by HD^-0.5); V stored transposed
// [bh,64,S] bf16 so the PV WMMA A-operand becomes contiguous row loads.
// ---------------------------------------------------------------------------
__global__ void __launch_bounds__(256) k_qkv_rope(
    const bf16_t* __restrict__ Xb,
    const bf16_t* __restrict__ Wqb, const bf16_t* __restrict__ Wkb,
    const bf16_t* __restrict__ Wvb,
    bf16_t* __restrict__ Qb, bf16_t* __restrict__ Kb, bf16_t* __restrict__ Vt) {
  const int lane = threadIdx.x & 31;
  const int wid  = threadIdx.x >> 5;
  const int half = lane >> 4;
  const int lq   = lane & 15;

  const int tile = blockIdx.x * 8 + wid;        // 128 rowblocks * 16 heads * 3
  const int rb   = tile / (H_ * 3);
  const int rem  = tile % (H_ * 3);
  const int h    = rem / 3;
  const int mat  = rem % 3;                     // 0=Q 1=K 2=V
  const int rowBase = rb * 32;

  const bf16_t* W = (mat == 0) ? Wqb : (mat == 1) ? Wkb : Wvb;

  const bf16_t* xrow0 = Xb + (long)(rowBase + lq) * D_;
  const bf16_t* xrow1 = xrow0 + 16 * D_;
  const bf16_t* wrow[4];
#pragma unroll
  for (int dt = 0; dt < 4; ++dt)
    wrow[dt] = W + (long)(h * 64 + dt * 16 + lq) * D_;

  v8f acc[2][4] = {};
  for (int k = 0; k < D_; k += 32) {
    __builtin_prefetch(xrow0 + k + 128, 0, 3);
    v16bf a0 = ldfrag(xrow0 + k + 8 * half, xrow0 + k + 16 + 8 * half);
    v16bf a1 = ldfrag(xrow1 + k + 8 * half, xrow1 + k + 16 + 8 * half);
#pragma unroll
    for (int dt = 0; dt < 4; ++dt) {
      v16bf b = ldfrag(wrow[dt] + k + 8 * half, wrow[dt] + k + 16 + 8 * half);
      acc[0][dt] = wmma_bf16(a0, b, acc[0][dt]);
      acc[1][dt] = wmma_bf16(a1, b, acc[1][dt]);
    }
  }

  const int b    = rowBase / S_;                // whole tile shares one batch
  const int sloc = rowBase % S_;
  const int bh   = b * H_ + h;

  if (mat == 2) {
    // V transposed store: lane owns column d = dt*16+lq; rows s contiguous in r
#pragma unroll
    for (int mg = 0; mg < 2; ++mg) {
#pragma unroll
      for (int dt = 0; dt < 4; ++dt) {
        bf16_t* dst = Vt + ((long)bh * 64 + dt * 16 + lq) * S_ + sloc + mg * 16 +
                      8 * half;
        uint4 u;
        u.x = pack2(acc[mg][dt][0], acc[mg][dt][1]);
        u.y = pack2(acc[mg][dt][2], acc[mg][dt][3]);
        u.z = pack2(acc[mg][dt][4], acc[mg][dt][5]);
        u.w = pack2(acc[mg][dt][6], acc[mg][dt][7]);
        *(uint4*)dst = u;
      }
    }
  } else {
    // RoPE: column d and d+32 live in the same lane (acc tiles dt and dt+2).
    // For Q additionally fold in the attention scale HD^-0.5 = 0.125.
    bf16_t* dstB = (mat == 0) ? Qb : Kb;
    const float qs  = (mat == 0) ? 0.125f : 1.0f;
    const float c32 = 13.287712379549449f / 32.0f;   // log2(10000)/32
    const float invA = exp2f(-((float)lq)        * c32);  // freq idx d%32 = lq
    const float invB = exp2f(-((float)(lq + 16)) * c32);  // freq idx lq+16
#pragma unroll
    for (int mg = 0; mg < 2; ++mg) {
#pragma unroll
      for (int r = 0; r < 8; ++r) {
        const int srow = sloc + mg * 16 + 8 * half + r;
        float sA, cA, sB, cB;
        __sincosf((float)srow * invA, &sA, &cA);
        __sincosf((float)srow * invB, &sB, &cB);
        const float a0 = acc[mg][0][r], a1 = acc[mg][1][r];
        const float a2 = acc[mg][2][r], a3 = acc[mg][3][r];
        const float o0 = (a0 * cA - a2 * sA) * qs;   // d = lq
        const float o1 = (a1 * cB - a3 * sB) * qs;   // d = 16+lq
        const float o2 = (a2 * cA + a0 * sA) * qs;   // d = 32+lq
        const float o3 = (a3 * cB + a1 * sB) * qs;   // d = 48+lq
        bf16_t* dr = dstB + ((long)bh * S_ + srow) * 64 + lq;
        dr[0]  = to_bf16(o0);
        dr[16] = to_bf16(o1);
        dr[32] = to_bf16(o2);
        dr[48] = to_bf16(o3);
      }
    }
  }
}

// ---------------------------------------------------------------------------
// Kernel 3: flash attention.  One wave owns 16 queries of one (b,h); the 8
// waves of a block are consecutive q-blocks of the SAME (b,h), so K/V tiles
// are shared through WGP$.  Computes S^T = K*Q^T tiles (softmax over keys is
// lane-local + one shfl_xor(16)), online softmax, O^T += V^T * P^T via WMMA.
// Q was pre-scaled by HD^-0.5 at projection time.
// ---------------------------------------------------------------------------
__global__ void __launch_bounds__(256) k_attn(
    const bf16_t* __restrict__ Qb, const bf16_t* __restrict__ Kb,
    const bf16_t* __restrict__ Vt, bf16_t* __restrict__ Ab) {
  const int lane = threadIdx.x & 31;
  const int wid  = threadIdx.x >> 5;
  const int half = lane >> 4;
  const int lq   = lane & 15;

  const int t  = blockIdx.x * 8 + wid;   // 32 bh * 128 qblocks
  const int qb = t & 127;
  const int bh = t >> 7;

  // Q B-operand (lane = query column, contiguous d along k-dim)
  const bf16_t* qrow = Qb + ((long)bh * S_ + qb * 16 + lq) * 64;
  const v16bf qf0 = ldfrag(qrow + 8 * half,      qrow + 16 + 8 * half);
  const v16bf qf1 = ldfrag(qrow + 32 + 8 * half, qrow + 48 + 8 * half);

  v8f   o[4] = {};
  float mrun = -3.0e38f, lsum = 0.0f;

  const bf16_t* kb0 = Kb + (long)bh * S_ * 64;
  const bf16_t* vb0 = Vt + (long)bh * 64 * S_;

  for (int kb = 0; kb < S_; kb += 32) {
    // K A-operand: lane = key row, k-dim = head dim (contiguous)
    const bf16_t* kr0 = kb0 + (long)(kb + lq) * 64;
    const bf16_t* kr1 = kr0 + 16 * 64;
    v16bf a00 = ldfrag(kr0 + 8 * half,      kr0 + 16 + 8 * half);
    v16bf a01 = ldfrag(kr0 + 32 + 8 * half, kr0 + 48 + 8 * half);
    v16bf a10 = ldfrag(kr1 + 8 * half,      kr1 + 16 + 8 * half);
    v16bf a11 = ldfrag(kr1 + 32 + 8 * half, kr1 + 48 + 8 * half);

    v8f c0 = {};                       // S^T rows = keys kb..kb+15
    c0 = wmma_bf16(a00, qf0, c0);
    c0 = wmma_bf16(a01, qf1, c0);
    v8f c1 = {};                       // keys kb+16..kb+31
    c1 = wmma_bf16(a10, qf0, c1);
    c1 = wmma_bf16(a11, qf1, c1);

    // per-query (lane-local) block max over all 32 keys of this step
    float lm = c0[0];
#pragma unroll
    for (int r = 1; r < 8; ++r) lm = fmaxf(lm, c0[r]);
#pragma unroll
    for (int r = 0; r < 8; ++r) lm = fmaxf(lm, c1[r]);
    lm = fmaxf(lm, __shfl_xor(lm, 16));

    const float mnew = fmaxf(mrun, lm);
    const float corr = __expf(mrun - mnew);

    float p0[8], p1[8];
    float bs = 0.0f;
#pragma unroll
    for (int r = 0; r < 8; ++r) {
      p0[r] = __expf(c0[r] - mnew);
      p1[r] = __expf(c1[r] - mnew);
      bs += p0[r] + p1[r];
    }
    bs += __shfl_xor(bs, 16);
    lsum = lsum * corr + bs;
    mrun = mnew;

#pragma unroll
    for (int dt = 0; dt < 4; ++dt) o[dt] = o[dt] * corr;

    // Pack P^T into B-operand: lane q needs k = 16*half + 0..15.
    // Own C rows cover k = {8*half..8*half+7} (tile0) and +16 (tile1);
    // the missing 8 come from lane q^16 via one shfl_xor per packed uint.
    unsigned u0[4], u1[4];
#pragma unroll
    for (int j = 0; j < 4; ++j) {
      u0[j] = pack2(p0[2 * j], p0[2 * j + 1]);
      u1[j] = pack2(p1[2 * j], p1[2 * j + 1]);
    }
    union { v16bf v; unsigned u[8]; } pb;
#pragma unroll
    for (int j = 0; j < 4; ++j) {
      const unsigned q0 = __shfl_xor(u0[j], 16);
      const unsigned q1 = __shfl_xor(u1[j], 16);
      pb.u[j]     = half ? q1    : u0[j];   // k = 16*half + 2j,2j+1
      pb.u[4 + j] = half ? u1[j] : q0;      // k = 16*half + 8+2j,...
    }

    // O^T += V^T(16d x 32k) * P^T(32k x 16q); V^T rows are contiguous in k
#pragma unroll
    for (int dt = 0; dt < 4; ++dt) {
      const bf16_t* vr = vb0 + (long)(dt * 16 + lq) * S_ + kb;
      v16bf va = ldfrag(vr + 16 * half, vr + 16 * half + 8);
      o[dt] = wmma_bf16(va, pb.v, o[dt]);
    }
  }

  const float invl = 1.0f / lsum;
  const int   b = bh >> 4, h = bh & 15;
  // O^T C-layout: lane column = query lq, rows d contiguous in r -> 16B stores
  bf16_t* orow = Ab + ((long)(b * S_ + qb * 16 + lq)) * D_ + h * 64 + 8 * half;
#pragma unroll
  for (int dt = 0; dt < 4; ++dt) {
    uint4 u;
    u.x = pack2(o[dt][0] * invl, o[dt][1] * invl);
    u.y = pack2(o[dt][2] * invl, o[dt][3] * invl);
    u.z = pack2(o[dt][4] * invl, o[dt][5] * invl);
    u.w = pack2(o[dt][6] * invl, o[dt][7] * invl);
    *(uint4*)(orow + dt * 16) = u;
  }
}

// ---------------------------------------------------------------------------
// Kernel 4: output projection  Out = Attn @ Wo^T  (f32 result).
// 32x64 wave tiles, same structure as the QKV GEMM.
// ---------------------------------------------------------------------------
__global__ void __launch_bounds__(256) k_oproj(
    const bf16_t* __restrict__ Ab, const bf16_t* __restrict__ Wob,
    float* __restrict__ Out) {
  const int lane = threadIdx.x & 31;
  const int wid  = threadIdx.x >> 5;
  const int half = lane >> 4;
  const int lq   = lane & 15;

  const int t  = blockIdx.x * 8 + wid;  // 128 rowblocks(32) * 16 colblocks(64)
  const int rb = t >> 4;
  const int cb = t & 15;
  const int rowBase = rb * 32;

  const bf16_t* arow0 = Ab + (long)(rowBase + lq) * D_;
  const bf16_t* arow1 = arow0 + 16 * D_;
  const bf16_t* wrow[4];
#pragma unroll
  for (int dt = 0; dt < 4; ++dt)
    wrow[dt] = Wob + (long)(cb * 64 + dt * 16 + lq) * D_;

  v8f acc[2][4] = {};
  for (int k = 0; k < D_; k += 32) {
    __builtin_prefetch(arow0 + k + 128, 0, 3);
    v16bf a0 = ldfrag(arow0 + k + 8 * half, arow0 + k + 16 + 8 * half);
    v16bf a1 = ldfrag(arow1 + k + 8 * half, arow1 + k + 16 + 8 * half);
#pragma unroll
    for (int dt = 0; dt < 4; ++dt) {
      v16bf b = ldfrag(wrow[dt] + k + 8 * half, wrow[dt] + k + 16 + 8 * half);
      acc[0][dt] = wmma_bf16(a0, b, acc[0][dt]);
      acc[1][dt] = wmma_bf16(a1, b, acc[1][dt]);
    }
  }
#pragma unroll
  for (int mg = 0; mg < 2; ++mg) {
#pragma unroll
    for (int dt = 0; dt < 4; ++dt) {
#pragma unroll
      for (int r = 0; r < 8; ++r) {
        Out[(long)(rowBase + mg * 16 + 8 * half + r) * D_ + cb * 64 + dt * 16 +
            lq] = acc[mg][dt][r];
      }
    }
  }
}

// ---------------------------------------------------------------------------
extern "C" void kernel_launch(void* const* d_in, const int* in_sizes, int n_in,
                              void* d_out, int out_size, void* d_ws,
                              size_t ws_size, hipStream_t stream) {
  (void)in_sizes; (void)n_in; (void)out_size; (void)ws_size;
  const float* X  = (const float*)d_in[0];
  const float* Wq = (const float*)d_in[1];
  const float* Wk = (const float*)d_in[2];
  const float* Wv = (const float*)d_in[3];
  const float* Wo = (const float*)d_in[4];
  float* Out = (float*)d_out;

  char* ws = (char*)d_ws;
  size_t off = 0;
  bf16_t* Xb  = (bf16_t*)(ws + off); off += (size_t)M_ * D_ * 2;   // 8 MB
  bf16_t* Wqb = (bf16_t*)(ws + off); off += (size_t)D_ * D_ * 2;   // 2 MB
  bf16_t* Wkb = (bf16_t*)(ws + off); off += (size_t)D_ * D_ * 2;
  bf16_t* Wvb = (bf16_t*)(ws + off); off += (size_t)D_ * D_ * 2;
  bf16_t* Wob = (bf16_t*)(ws + off); off += (size_t)D_ * D_ * 2;
  bf16_t* Qb  = (bf16_t*)(ws + off); off += (size_t)M_ * D_ * 2;   // [bh,S,64]
  bf16_t* Kb  = (bf16_t*)(ws + off); off += (size_t)M_ * D_ * 2;   // [bh,S,64]
  bf16_t* Vt  = (bf16_t*)(ws + off); off += (size_t)M_ * D_ * 2;   // [bh,64,S]
  bf16_t* Ab  = (bf16_t*)(ws + off); off += (size_t)M_ * D_ * 2;   // [M,D]
  // total 48 MB of workspace

  const long nprep = (long)M_ * D_ + 4L * D_ * D_;                 // 8388608
  k_prep<<<(int)((nprep + 255) / 256), 256, 0, stream>>>(
      X, Wq, Wk, Wv, Wo, Xb, Wqb, Wkb, Wvb, Wob);

  // 128 rowblocks(32) * 16 heads * 3 mats = 6144 wave-tiles / 8 per block
  k_qkv_rope<<<768, 256, 0, stream>>>(Xb, Wqb, Wkb, Wvb, Qb, Kb, Vt);

  // 32 (b,h) * 128 q-blocks = 4096 wave-tiles / 8 waves per block
  k_attn<<<512, 256, 0, stream>>>(Qb, Kb, Vt, Ab);

  // 128 rowblocks(32) * 16 colblocks = 2048 wave-tiles / 8 waves per block
  k_oproj<<<256, 256, 0, stream>>>(Ab, Wob, Out);
}
